// LSTMCell_19043884990893
// MI455X (gfx1250) — compile-verified, matrix operand order checked by reading, not run on hardware
//
#include <hip/hip_runtime.h>

// ---- WMMA vector types (CDNA5 / gfx1250, wave32) ----
typedef __attribute__((ext_vector_type(16))) __bf16 v16bf;
typedef __attribute__((ext_vector_type(8)))  __bf16 v8bf;
typedef __attribute__((ext_vector_type(8)))  float  v8f;
typedef __attribute__((ext_vector_type(2)))  float  v2f;
typedef __attribute__((ext_vector_type(2)))  __bf16 v2bf;

#define B_DIM 8192
#define K_DIM 1024
#define H_DIM 1024

#define BM 128         // rows (batch) per block
#define BN 64          // cols (hidden) per block
#define KC 32          // K-chunk per WMMA step (bf16 K=32)
#define LDT 40         // padded LDS row stride in bf16 elems (80B: bank-stride 20, 16B aligned)
#define NTHREADS 512   // 16 waves, 4M x 4N
#define KSTEPS ((K_DIM + H_DIM) / KC)   // 64

// float2 -> packed bf16x2 via v_cvt_pk_bf16_f32 (1 VALU op, RNE)
__device__ __forceinline__ unsigned pack2(float a, float b) {
  v2f f; f.x = a; f.y = b;
  v2bf r = __builtin_convertvector(f, v2bf);
  return __builtin_bit_cast(unsigned, r);
}
__device__ __forceinline__ uint2 pack4(float4 v) {
  uint2 r;
  r.x = pack2(v.x, v.y);
  r.y = pack2(v.z, v.w);
  return r;
}
__device__ __forceinline__ float fast_sigmoid(float x) {
  return 1.0f / (1.0f + __expf(-x));
}
__device__ __forceinline__ float fast_tanh(float x) {
  float e = __expf(2.0f * x);
  return (e - 1.0f) / (e + 1.0f);
}

// CDNA5 LDS-only workgroup barrier: orders DS ops across the workgroup but
// does NOT drain LOADcnt, so prefetched global loads stay in flight across it.
__device__ __forceinline__ void wg_barrier_lds() {
  asm volatile(
      "s_wait_dscnt 0x0\n\t"
      "s_barrier_signal -1\n\t"
      "s_barrier_wait -1"
      ::: "memory");
}

union Frag16 { v16bf v; v8bf h[2]; };

__device__ __forceinline__ v8f wmma_bf16(const Frag16& a, const Frag16& b, v8f c) {
  return __builtin_amdgcn_wmma_f32_16x16x32_bf16(
      false, a.v, false, b.v, (short)0, c, false, false);
}

__global__ __launch_bounds__(NTHREADS)
void lstm_cell_wmma(const float* __restrict__ x,
                    const float* __restrict__ cin,
                    const float* __restrict__ hin,
                    const float* __restrict__ Wii, const float* __restrict__ Wif,
                    const float* __restrict__ Wig, const float* __restrict__ Wio,
                    const float* __restrict__ Whi, const float* __restrict__ Whf,
                    const float* __restrict__ Whg, const float* __restrict__ Who,
                    const float* __restrict__ bi,  const float* __restrict__ bf,
                    const float* __restrict__ bg,  const float* __restrict__ bo,
                    float* __restrict__ out_c, float* __restrict__ out_h)
{
  // double-buffered tiles: 2 * (10 KB + 20 KB) = 60 KB of 320 KB/WGP
  __shared__ __align__(16) unsigned short sA[2][BM * LDT];
  __shared__ __align__(16) unsigned short sB[2][4][BN * LDT];

  const int tid   = threadIdx.x;
  const int lane  = tid & 31;
  const int wave  = tid >> 5;          // 16 waves
  const int wm    = wave >> 2;         // 0..3  (32-row slab)
  const int wn    = wave & 3;          // 0..3  (16-col slab)
  const int lrow  = lane & 15;         // M (A) / N (B) index within 16
  const int lksel = lane >> 4;         // K phase select (0 or 1)
  const int bn0   = blockIdx.x * BN;
  const int bm0   = blockIdx.y * BM;

  // ---- thread-invariant staging offsets (computed once) ----
  // A: two float4 quads per thread: rows (tid>>3) and (tid>>3)+64, cols (tid&7)*4
  const int    am    = tid >> 3;
  const int    ak    = (tid & 7) * 4;
  const size_t aGoff = (size_t)(bm0 + am) * K_DIM + ak;   // + k0 per step
  const int    aSoff = am * LDT + ak;                     // second quad: +64*LDT
  // B: 2x2 patch per thread per gate: n-pair (tid&31)*2, k-pair (tid>>5)*2
  const int    bnn   = (tid & 31) * 2;
  const int    bkk   = (tid >> 5) * 2;
  const size_t wGoff = (size_t)bkk * H_DIM + bn0 + bnn;   // + k0*H per step
  const int    bSoff = bnn * LDT + bkk;

  // ---- thread-invariant fragment offsets ----
  const int aFoff0 = (wm * 32 + lrow) * LDT + lksel * 8;  // m-tile 0
  const int aFoff1 = aFoff0 + 16 * LDT;                   // m-tile 1
  const int bFoff  = (wn * 16 + lrow) * LDT + lksel * 8;

  const v8f vzero = {};
  v8f acc[4][2];                      // [gate][m-tile]
  #pragma unroll
  for (int g = 0; g < 4; ++g) {
    acc[g][0] = vzero;
    acc[g][1] = vzero;
  }

  // ---- register stage for the global->LDS pipeline (holds tile kt+1) ----
  float4 ra0, ra1;
  v2f rb0[4], rb1[4];

  auto fetch = [&](const float* __restrict__ A,
                   const float* __restrict__ V0, const float* __restrict__ V1,
                   const float* __restrict__ V2, const float* __restrict__ V3,
                   int k0) {
    ra0 = *(const float4*)(A + aGoff + k0);
    ra1 = *(const float4*)(A + aGoff + (size_t)64 * K_DIM + k0);
    const size_t wk = (size_t)k0 * H_DIM + wGoff;
    rb0[0] = *(const v2f*)(V0 + wk);  rb1[0] = *(const v2f*)(V0 + wk + H_DIM);
    rb0[1] = *(const v2f*)(V1 + wk);  rb1[1] = *(const v2f*)(V1 + wk + H_DIM);
    rb0[2] = *(const v2f*)(V2 + wk);  rb1[2] = *(const v2f*)(V2 + wk + H_DIM);
    rb0[3] = *(const v2f*)(V3 + wk);  rb1[3] = *(const v2f*)(V3 + wk + H_DIM);
  };
  auto store_stage = [&](int buf) {
    *(uint2*)(&sA[buf][aSoff])            = pack4(ra0);
    *(uint2*)(&sA[buf][aSoff + 64 * LDT]) = pack4(ra1);
    #pragma unroll
    for (int g = 0; g < 4; ++g) {
      *(unsigned*)(&sB[buf][g][bSoff])       = pack2(rb0[g].x, rb1[g].x);
      *(unsigned*)(&sB[buf][g][bSoff + LDT]) = pack2(rb0[g].y, rb1[g].y);
    }
  };

  // ---- prologue: tile0 -> LDS buf0; tile1 fetched into regs ----
  fetch(x, Wii, Wif, Wig, Wio, 0);
  store_stage(0);
  fetch(x, Wii, Wif, Wig, Wio, KC);
  wg_barrier_lds();

  // phase unrolled at compile time -> pointer selection folds to constants
  #pragma unroll
  for (int phase = 0; phase < 2; ++phase) {
    const float* A  = phase ? hin : x;
    const float* V0 = phase ? Whi : Wii;
    const float* V1 = phase ? Whf : Wif;
    const float* V2 = phase ? Whg : Wig;
    const float* V3 = phase ? Who : Wio;

    for (int kc = 0; kc < 32; ++kc) {
      const int kt  = (phase << 5) | kc;
      const int cur = kt & 1;

      // 1) drain stage regs (tile kt+1, fetched a full iteration ago) into
      //    the other LDS buffer; safe: that buffer was last read at kt-1.
      if (!(phase == 1 && kc == 31)) store_stage((kt + 1) & 1);

      // 2) issue global loads for tile kt+2: a whole iteration of cover
      //    (they now survive the barrier un-waited).
      if (phase == 0) {
        if (kc < 30)       fetch(A, V0, V1, V2, V3, (kc + 2) * KC);
        else if (kc == 30) fetch(hin, Whi, Whf, Whg, Who, 0);
        else               fetch(hin, Whi, Whf, Whg, Who, KC);
      } else if (kc < 30) {
        fetch(A, V0, V1, V2, V3, (kc + 2) * KC);
      }

      // 3) A fragments (ISA 16-bit 16x32 striping), reused across 4 gates
      Frag16 a0, a1;
      {
        const unsigned short* pa0 = &sA[cur][aFoff0];
        const unsigned short* pa1 = &sA[cur][aFoff1];
        a0.h[0] = *(const v8bf*)(pa0); a0.h[1] = *(const v8bf*)(pa0 + 16);
        a1.h[0] = *(const v8bf*)(pa1); a1.h[1] = *(const v8bf*)(pa1 + 16);
      }

      // 4) 4 gates: each B fragment feeds both m-tiles (2 WMMAs)
      Frag16 bq[4];
      #pragma unroll
      for (int g = 0; g < 4; ++g) {
        const unsigned short* pb = &sB[cur][g][bFoff];
        bq[g].h[0] = *(const v8bf*)(pb);
        bq[g].h[1] = *(const v8bf*)(pb + 16);
      }
      #pragma unroll
      for (int g = 0; g < 4; ++g) {
        acc[g][0] = wmma_bf16(a0, bq[g], acc[g][0]);
        acc[g][1] = wmma_bf16(a1, bq[g], acc[g][1]);
      }

      wg_barrier_lds();
    }
  }

  // ---- fused epilogue: gates + cell update, straight from accumulators ----
  // C/D layout: VGPR j -> M = j + 8*(lane>=16), N = lane%16
  const int col = bn0 + wn * 16 + lrow;
  const float bii = bi[col], bff = bf[col], bgg = bg[col], boo = bo[col];
  #pragma unroll
  for (int mt = 0; mt < 2; ++mt) {
    const int mbase = bm0 + wm * 32 + mt * 16 + lksel * 8;
    #pragma unroll
    for (int j = 0; j < 8; ++j) {
      const size_t idx = (size_t)(mbase + j) * H_DIM + col;
      const float cv = cin[idx];
      const float iv = fast_sigmoid(acc[0][mt][j] + bii);
      const float fv = fast_sigmoid(acc[1][mt][j] + bff);
      const float gv = fast_tanh   (acc[2][mt][j] + bgg);
      const float ov = fast_sigmoid(acc[3][mt][j] + boo);
      const float nc = fv * cv + iv * gv;
      out_c[idx] = nc;
      out_h[idx] = ov * fast_tanh(nc);
    }
  }
}

extern "C" void kernel_launch(void* const* d_in, const int* in_sizes, int n_in,
                              void* d_out, int out_size, void* d_ws, size_t ws_size,
                              hipStream_t stream) {
  const float* x   = (const float*)d_in[0];
  const float* c   = (const float*)d_in[1];
  const float* h   = (const float*)d_in[2];
  const float* Wii = (const float*)d_in[3];
  const float* Wif = (const float*)d_in[4];
  const float* Wig = (const float*)d_in[5];
  const float* Wio = (const float*)d_in[6];
  const float* Whi = (const float*)d_in[7];
  const float* Whf = (const float*)d_in[8];
  const float* Whg = (const float*)d_in[9];
  const float* Who = (const float*)d_in[10];
  const float* bi  = (const float*)d_in[11];
  const float* bf  = (const float*)d_in[12];
  const float* bg  = (const float*)d_in[13];
  const float* bo  = (const float*)d_in[14];

  float* out_c = (float*)d_out;
  float* out_h = out_c + (size_t)B_DIM * H_DIM;

  dim3 grid(H_DIM / BN, B_DIM / BM);   // 16 x 64 blocks
  lstm_cell_wmma<<<grid, NTHREADS, 0, stream>>>(x, c, h,
                                                Wii, Wif, Wig, Wio,
                                                Whi, Whf, Whg, Who,
                                                bi, bf, bg, bo,
                                                out_c, out_h);
}
